// MaskedMultiHeadAttention_19353122635845
// MI455X (gfx1250) — matrix operand order repaired and checked by reading, not compiled, not run on hardware
//
#include <hip/hip_runtime.h>
#include <math.h>

typedef __bf16 bf16_t;
typedef bf16_t v16bf __attribute__((ext_vector_type(16)));
typedef bf16_t v8bf  __attribute__((ext_vector_type(8)));
typedef float  v8f   __attribute__((ext_vector_type(8)));

#define NB   256                 // batch
#define NT   200                 // seq len
#define NE   584                 // embed dim
#define NH   8                   // heads
#define NHS  73                  // head size
#define NM   (NB*NT)             // 51200 rows
#define KP   608                 // NE padded to mult of 32 (K dim of GEMMs)
#define NP   640                 // NE padded to mult of 64 (weight rows)
#define TP   208                 // NT padded to mult of 16
#define EP   96                  // head size padded to mult of 32 (scores K dim)
#define FP   80                  // head size padded to mult of 16 (output N dim)
#define TP2  224                 // key/time padding for transposed V columns

static __device__ __forceinline__ bf16_t f2bf(float f) {
    unsigned int u = __float_as_uint(f);
    u += 0x7FFFu + ((u >> 16) & 1u);            // round-to-nearest-even
    unsigned short s = (unsigned short)(u >> 16);
    return __builtin_bit_cast(bf16_t, s);
}

// A-operand (16-bit 16x32, ISA 7.12.2): lane half owns K {h*8..h*8+7, h*8+16..h*8+23}
static __device__ __forceinline__ v16bf load_afrag(const bf16_t* p) {
    v8bf lo = *(const v8bf*)(p);
    v8bf hi = *(const v8bf*)(p + 16);
    return __builtin_shufflevector(lo, hi, 0,1,2,3,4,5,6,7,8,9,10,11,12,13,14,15);
}
// B-operand (16-bit 32x16): lane half owns K {h*8..h*8+7, h*8+16..h*8+23}, same as A
static __device__ __forceinline__ v16bf load_bfrag(const bf16_t* p) {
    v8bf lo = *(const v8bf*)(p);
    v8bf hi = *(const v8bf*)(p + 16);
    return __builtin_shufflevector(lo, hi, 0,1,2,3,4,5,6,7,8,9,10,11,12,13,14,15);
}

#define WMMA_BF16(A, Bv, C) \
    __builtin_amdgcn_wmma_f32_16x16x32_bf16(false, (A), false, (Bv), (short)0, (C), false, false)

// ---------------------------------------------------------------- zero init
__global__ void zero_kernel(uint4* __restrict__ p, size_t n16) {
    size_t i = (size_t)blockIdx.x * blockDim.x + threadIdx.x;
    size_t stride = (size_t)gridDim.x * blockDim.x;
    uint4 z = {0u, 0u, 0u, 0u};
    for (; i < n16; i += stride) p[i] = z;
}

// ------------------------------------------------------- fp32 -> bf16 packs
__global__ void convert_x_kernel(const float* __restrict__ x, bf16_t* __restrict__ xb) {
    size_t i = (size_t)blockIdx.x * 256 + threadIdx.x;
    if (i >= (size_t)NM * KP) return;
    int    k = (int)(i % KP);
    size_t m = i / KP;
    float v = (k < NE) ? x[m * NE + k] : 0.0f;
    xb[i] = f2bf(v);
}

__global__ void convert_w_kernel(const float* __restrict__ Wq, const float* __restrict__ Wk,
                                 const float* __restrict__ Wv, const float* __restrict__ Wo,
                                 bf16_t* __restrict__ wb) {
    size_t i = (size_t)blockIdx.x * 256 + threadIdx.x;
    if (i >= (size_t)4 * NP * KP) return;
    int k = (int)(i % KP);
    int r = (int)((i / KP) % NP);
    int g = (int)(i / ((size_t)NP * KP));
    const float* W = (g == 0) ? Wq : (g == 1) ? Wk : (g == 2) ? Wv : Wo;
    float v = (r < NE && k < NE) ? W[(size_t)r * NE + k] : 0.0f;
    wb[i] = f2bf(v);
}

// -------------------------------------------------------------- QKV GEMM
// C[m,n] = sum_k xb[m,k] * W[n,k]; wave computes 32Mx64N: 2 A-frags x 4 B-frags
// -> 8 WMMAs per 12 VMEM loads per k-step (B frags reused across 2 WMMAs).
__global__ void __launch_bounds__(256) gemm_qkv_kernel(const bf16_t* __restrict__ xb,
                                                       const bf16_t* __restrict__ wb3,
                                                       bf16_t* __restrict__ qb,
                                                       bf16_t* __restrict__ kb,
                                                       bf16_t* __restrict__ vt) {
    const int lane = threadIdx.x & 31;
    const int wave = threadIdx.x >> 5;
    const int half = lane >> 4;
    const int l16  = lane & 15;
    const int m0   = (blockIdx.x * 8 + wave) * 32;
    const int n0   = blockIdx.y * 64;
    const int qkv  = blockIdx.z;

    const bf16_t* Aw0 = xb + (size_t)(m0 + l16) * KP + half * 8;
    const bf16_t* Aw1 = Aw0 + (size_t)16 * KP;
    const bf16_t* Bw  = wb3 + (size_t)qkv * NP * KP + (size_t)(n0 + l16) * KP + half * 8;

    v8f acc[2][4];
    v8f zf = {0.f,0.f,0.f,0.f,0.f,0.f,0.f,0.f};
    #pragma unroll
    for (int i = 0; i < 2; ++i)
        #pragma unroll
        for (int j = 0; j < 4; ++j) acc[i][j] = zf;

    for (int k = 0; k < KP; k += 32) {
        __builtin_prefetch(Aw0 + k + 64, 0, 0);          // global_prefetch_b8
        __builtin_prefetch(Aw1 + k + 64, 0, 0);
        v16bf a0 = load_afrag(Aw0 + k);
        v16bf a1 = load_afrag(Aw1 + k);
        v16bf b0 = load_bfrag(Bw + k);
        v16bf b1 = load_bfrag(Bw + (size_t)16 * KP + k);
        v16bf b2 = load_bfrag(Bw + (size_t)32 * KP + k);
        v16bf b3 = load_bfrag(Bw + (size_t)48 * KP + k);
        acc[0][0] = WMMA_BF16(a0, b0, acc[0][0]);
        acc[1][0] = WMMA_BF16(a1, b0, acc[1][0]);
        acc[0][1] = WMMA_BF16(a0, b1, acc[0][1]);
        acc[1][1] = WMMA_BF16(a1, b1, acc[1][1]);
        acc[0][2] = WMMA_BF16(a0, b2, acc[0][2]);
        acc[1][2] = WMMA_BF16(a1, b2, acc[1][2]);
        acc[0][3] = WMMA_BF16(a0, b3, acc[0][3]);
        acc[1][3] = WMMA_BF16(a1, b3, acc[1][3]);
    }

    const float qscale = 0.041380294f;                  // 584^-0.5 folded into Q
    #pragma unroll
    for (int mi = 0; mi < 2; ++mi) {
        #pragma unroll
        for (int nn = 0; nn < 4; ++nn) {
            int col = n0 + nn * 16 + l16;
            if (col >= NE) continue;
            int h = col / NHS, f = col % NHS;
            #pragma unroll
            for (int vr = 0; vr < 8; ++vr) {
                int m = m0 + mi * 16 + vr + 8 * half;
                int b = m / NT, t = m % NT;
                float v = acc[mi][nn][vr];
                if (qkv == 0) {
                    qb[(((size_t)b * NH + h) * TP + t) * EP + f] = f2bf(v * qscale);
                } else if (qkv == 1) {
                    kb[(((size_t)b * NH + h) * TP + t) * EP + f] = f2bf(v);
                } else {    // V stored transposed: [b][h][f][t]
                    vt[(((size_t)b * NH + h) * FP + f) * TP2 + t] = f2bf(v);
                }
            }
        }
    }
}

// ---------------------------------------------------- flash causal attention
// One wave handles 16 query rows of one (b,h). Online softmax in C layout.
__global__ void __launch_bounds__(32) attn_kernel(const bf16_t* __restrict__ qb,
                                                  const bf16_t* __restrict__ kb,
                                                  const bf16_t* __restrict__ vt,
                                                  bf16_t* __restrict__ ob) {
    const int bh   = blockIdx.x;            // b*NH + h
    const int r    = blockIdx.y;            // query row tile
    const int lane = threadIdx.x & 31;
    const int half = lane >> 4;
    const int l16  = lane & 15;

    const bf16_t* qbase = qb + (size_t)bh * TP * EP;
    const bf16_t* kbase = kb + (size_t)bh * TP * EP;
    const bf16_t* vbase = vt + (size_t)bh * FP * TP2;

    __shared__ __align__(32) bf16_t lds[16 * 32];
    {   // zero LDS tile once: upper K half (16..31) stays zero so P fits K=32 WMMA
        bf16_t bz = f2bf(0.0f);
        for (int i = threadIdx.x; i < 16 * 32; i += 32) lds[i] = bz;
    }
    __syncthreads();

    v16bf aq[3];
    #pragma unroll
    for (int ks = 0; ks < 3; ++ks)
        aq[ks] = load_afrag(qbase + (size_t)(r * 16 + l16) * EP + ks * 32 + half * 8);

    float mrun[8], lrun[8];
    v8f o[5];
    v8f zf = {0.f,0.f,0.f,0.f,0.f,0.f,0.f,0.f};
    #pragma unroll
    for (int i = 0; i < 8; ++i) { mrun[i] = -INFINITY; lrun[i] = 0.0f; }
    #pragma unroll
    for (int i = 0; i < 5; ++i) o[i] = zf;

    for (int c = 0; c <= r; ++c) {
        // S = Q K^T   (Q already scaled by E^-0.5)
        v8f s = zf;
        #pragma unroll
        for (int ks = 0; ks < 3; ++ks) {
            v16bf bk = load_bfrag(kbase + (size_t)(c * 16 + l16) * EP + ks * 32 + half * 8);
            s = WMMA_BF16(aq[ks], bk, s);
        }
        const int col = c * 16 + l16;
        float p[8];
        #pragma unroll
        for (int vr = 0; vr < 8; ++vr) {
            int row = r * 16 + vr + 8 * half;
            float sv = s[vr];
            if (c == r && col > row) sv = -INFINITY;        // causal mask (covers t>=NT too)
            float rmax = sv;                                 // row max across the 16-lane half
            rmax = fmaxf(rmax, __shfl_xor(rmax, 1, 32));
            rmax = fmaxf(rmax, __shfl_xor(rmax, 2, 32));
            rmax = fmaxf(rmax, __shfl_xor(rmax, 4, 32));
            rmax = fmaxf(rmax, __shfl_xor(rmax, 8, 32));
            float mnew  = fmaxf(mrun[vr], rmax);
            float alpha = __expf(mrun[vr] - mnew);
            float pv    = __expf(sv - mnew);
            float rsum  = pv;
            rsum += __shfl_xor(rsum, 1, 32);
            rsum += __shfl_xor(rsum, 2, 32);
            rsum += __shfl_xor(rsum, 4, 32);
            rsum += __shfl_xor(rsum, 8, 32);
            lrun[vr] = lrun[vr] * alpha + rsum;
            mrun[vr] = mnew;
            p[vr] = pv;
            #pragma unroll
            for (int ft = 0; ft < 5; ++ft) o[ft][vr] *= alpha;
        }
        // transpose P (C layout -> A layout) through LDS
        #pragma unroll
        for (int vr = 0; vr < 8; ++vr)
            lds[(vr + 8 * half) * 32 + l16] = f2bf(p[vr]);
        __syncthreads();
        v16bf pa = load_afrag(&lds[l16 * 32 + half * 8]);   // K 16..31 are the zeroed half
        __syncthreads();
        // O += P V  (V transposed: B loads contiguous over t)
        #pragma unroll
        for (int ft = 0; ft < 5; ++ft) {
            v16bf bv = load_bfrag(vbase + (size_t)(ft * 16 + l16) * TP2 + c * 16 + half * 8);
            o[ft] = WMMA_BF16(pa, bv, o[ft]);
        }
    }

    const int b = bh >> 3, h = bh & 7;
    #pragma unroll
    for (int ft = 0; ft < 5; ++ft) {
        int f = ft * 16 + l16;
        if (f >= NHS) continue;
        #pragma unroll
        for (int vr = 0; vr < 8; ++vr) {
            int t = r * 16 + vr + 8 * half;
            if (t >= NT) continue;
            float v = o[ft][vr] / lrun[vr];
            ob[((size_t)b * NT + t) * KP + h * NHS + f] = f2bf(v);
        }
    }
}

// ------------------------------------------------------------ output GEMM
__global__ void __launch_bounds__(256) gemm_out_kernel(const bf16_t* __restrict__ ob,
                                                       const bf16_t* __restrict__ wob,
                                                       const float* __restrict__ bo,
                                                       float* __restrict__ out) {
    const int lane = threadIdx.x & 31;
    const int wave = threadIdx.x >> 5;
    const int half = lane >> 4;
    const int l16  = lane & 15;
    const int m0   = (blockIdx.x * 8 + wave) * 32;
    const int n0   = blockIdx.y * 64;

    const bf16_t* Aw0 = ob + (size_t)(m0 + l16) * KP + half * 8;
    const bf16_t* Aw1 = Aw0 + (size_t)16 * KP;
    const bf16_t* Bw  = wob + (size_t)(n0 + l16) * KP + half * 8;

    v8f acc[2][4];
    v8f zf = {0.f,0.f,0.f,0.f,0.f,0.f,0.f,0.f};
    #pragma unroll
    for (int i = 0; i < 2; ++i)
        #pragma unroll
        for (int j = 0; j < 4; ++j) acc[i][j] = zf;

    for (int k = 0; k < KP; k += 32) {
        __builtin_prefetch(Aw0 + k + 64, 0, 0);
        __builtin_prefetch(Aw1 + k + 64, 0, 0);
        v16bf a0 = load_afrag(Aw0 + k);
        v16bf a1 = load_afrag(Aw1 + k);
        v16bf b0 = load_bfrag(Bw + k);
        v16bf b1 = load_bfrag(Bw + (size_t)16 * KP + k);
        v16bf b2 = load_bfrag(Bw + (size_t)32 * KP + k);
        v16bf b3 = load_bfrag(Bw + (size_t)48 * KP + k);
        acc[0][0] = WMMA_BF16(a0, b0, acc[0][0]);
        acc[1][0] = WMMA_BF16(a1, b0, acc[1][0]);
        acc[0][1] = WMMA_BF16(a0, b1, acc[0][1]);
        acc[1][1] = WMMA_BF16(a1, b1, acc[1][1]);
        acc[0][2] = WMMA_BF16(a0, b2, acc[0][2]);
        acc[1][2] = WMMA_BF16(a1, b2, acc[1][2]);
        acc[0][3] = WMMA_BF16(a0, b3, acc[0][3]);
        acc[1][3] = WMMA_BF16(a1, b3, acc[1][3]);
    }

    #pragma unroll
    for (int mi = 0; mi < 2; ++mi) {
        #pragma unroll
        for (int nn = 0; nn < 4; ++nn) {
            int col = n0 + nn * 16 + l16;
            if (col >= NE) continue;
            float bias = bo[col];
            #pragma unroll
            for (int vr = 0; vr < 8; ++vr) {
                int m = m0 + mi * 16 + vr + 8 * half;
                out[(size_t)m * NE + col] = acc[mi][nn][vr] + bias;
            }
        }
    }
}

// ---------------------------------------------------------------- launcher
extern "C" void kernel_launch(void* const* d_in, const int* in_sizes, int n_in,
                              void* d_out, int out_size, void* d_ws, size_t ws_size,
                              hipStream_t stream) {
    (void)in_sizes; (void)n_in; (void)out_size; (void)ws_size;
    const float* x  = (const float*)d_in[0];
    const float* Wq = (const float*)d_in[1];
    const float* Wk = (const float*)d_in[2];
    const float* Wv = (const float*)d_in[3];
    const float* Wo = (const float*)d_in[4];
    const float* bo = (const float*)d_in[5];
    float* out = (float*)d_out;

    char* ws = (char*)d_ws;
    const size_t QB = (size_t)NB * NH * TP * EP * 2;     // 81,788,928
    const size_t KBsz = QB;
    const size_t VT = (size_t)NB * NH * FP * TP2 * 2;    // 73,400,320
    const size_t OB = (size_t)NM * KP * 2;               // 62,259,200
    const size_t XB = (size_t)NM * KP * 2;

    bf16_t* qb  = (bf16_t*)(ws);
    bf16_t* kb  = (bf16_t*)(ws + QB);
    bf16_t* vt  = (bf16_t*)(ws + QB + KBsz);
    bf16_t* ob  = (bf16_t*)(ws + QB + KBsz + VT);
    bf16_t* xb  = (bf16_t*)(ws + QB + KBsz + VT + OB);
    bf16_t* wb  = (bf16_t*)(ws + QB + KBsz + VT + OB + XB);   // [4][NP][KP]: Wq,Wk,Wv,Wo
    bf16_t* wob = wb + (size_t)3 * NP * KP;

    // zero the padded q/k/vt/ob region (so padding is exact zero in WMMA inputs)
    size_t zero16 = (QB + KBsz + VT + OB) / 16;
    zero_kernel<<<2048, 256, 0, stream>>>((uint4*)ws, zero16);

    size_t nx = (size_t)NM * KP;
    convert_x_kernel<<<(unsigned)((nx + 255) / 256), 256, 0, stream>>>(x, xb);
    size_t nw = (size_t)4 * NP * KP;
    convert_w_kernel<<<(unsigned)((nw + 255) / 256), 256, 0, stream>>>(Wq, Wk, Wv, Wo, wb);

    dim3 g1(NM / 32 / 8, NP / 64, 3);                    // (200, 10, 3)
    gemm_qkv_kernel<<<g1, 256, 0, stream>>>(xb, wb, qb, kb, vt);

    dim3 g2(NB * NH, TP / 16, 1);                        // (2048, 13)
    attn_kernel<<<g2, 32, 0, stream>>>(qb, kb, vt, ob);

    dim3 g3(NM / 32 / 8, NP / 64, 1);                    // (200, 10)
    gemm_out_kernel<<<g3, 256, 0, stream>>>(ob, wob, bo, out);
}
